// EPAN_3d_8735963480602
// MI455X (gfx1250) — matrix operand order repaired and checked by reading, not compiled, not run on hardware
//
#include <hip/hip_runtime.h>

#define B_ 2
#define C_ 256
#define S_ 24
#define N_ (S_*S_*S_)   // 13824
#define NH_ 4
#define DH_ 64
#define P_ 64
#define EPS 1e-5f

typedef __attribute__((ext_vector_type(16))) _Float16 v16h;
typedef __attribute__((ext_vector_type(8)))  float    v8f;

// ---------------------------------------------------------------- WMMA helpers
__device__ __forceinline__ v8f wmma_f16(v16h a, v16h b, v8f c) {
  return __builtin_amdgcn_wmma_f32_16x16x32_f16(false, a, false, b, (short)0, c,
                                                false, false);
}

// A-matrix 16x32 (MxK) f16 layout: lane<16 -> row=lane, K={0..7,16..23};
// lane>=16 -> row=lane-16, K={8..15,24..31}
template <typename F>
__device__ __forceinline__ v16h load_a_gen(F f, int lane) {
  int m  = lane & 15;
  int k0 = (lane < 16) ? 0 : 8;
  int k1 = (lane < 16) ? 16 : 24;
  v16h a;
#pragma unroll
  for (int j = 0; j < 8; ++j) a[j] = (_Float16)f(m, k0 + j);
#pragma unroll
  for (int j = 0; j < 8; ++j) a[8 + j] = (_Float16)f(m, k1 + j);
  return a;
}

// B-matrix 32x16 (KxN): lane = K row, 16 halfs = the 16 columns
template <typename F>
__device__ __forceinline__ v16h load_b_gen(F f, int lane) {
  v16h b;
#pragma unroll
  for (int j = 0; j < 16; ++j) b[j] = (_Float16)f(lane, j);
  return b;
}

__device__ __forceinline__ v16h load_a_h(const _Float16* t, int ld, int lane) {
  return load_a_gen([&](int m, int k) { return t[m * ld + k]; }, lane);
}
__device__ __forceinline__ v16h load_b_h(const _Float16* t, int ld, int lane) {
  v16h b;
  int n  = lane & 15;
  int k0 = (lane < 16) ? 0 : 8;
#pragma unroll
  for (int j = 0; j < 16; ++j) b[j] = t[(k0 + (j < 8 ? j : j + 8)) * ld + n];
  return b;
}

__device__ __forceinline__ float block_reduce_sum(float v, float* sm) {
  int tid = threadIdx.x;
  sm[tid] = v;
  __syncthreads();
  for (int s = blockDim.x >> 1; s > 0; s >>= 1) {
    if (tid < s) sm[tid] += sm[tid + s];
    __syncthreads();
  }
  float r = sm[0];
  __syncthreads();
  return r;
}

// ----------------------------------------------- f32 -> f16 weight pre-convert
__global__ void cvt_f16_kernel(const float* __restrict__ src,
                               _Float16* __restrict__ dst, int n) {
  int i = blockIdx.x * 256 + threadIdx.x;
  if (i < n) dst[i] = (_Float16)src[i];
}

// ------------------------------------------------- grouped conv + instance norm
__global__ void gconv_inorm_kernel(const float* __restrict__ in,
                                   const float* __restrict__ w,
                                   const float* __restrict__ bias,
                                   const float* __restrict__ res,
                                   float* __restrict__ out) {
  __shared__ float red[256];
  int bc = blockIdx.x;
  int b = bc / C_, c = bc % C_;
  int g = c >> 5, co = c & 31;
  float wreg[32];
#pragma unroll
  for (int ci = 0; ci < 32; ++ci) wreg[ci] = w[(g * 32 + ci) * 32 + co];
  const float* inb = in + ((size_t)b * C_ + g * 32) * N_;
  float bv = bias[c];
  float* outr = out + (size_t)bc * N_;
  float s = 0.f, s2 = 0.f;
  for (int n = threadIdx.x; n < N_; n += 256) {
    float acc = bv;
#pragma unroll
    for (int ci = 0; ci < 32; ++ci) acc += inb[(size_t)ci * N_ + n] * wreg[ci];
    outr[n] = acc;
    s += acc; s2 += acc * acc;
  }
  s  = block_reduce_sum(s, red);
  s2 = block_reduce_sum(s2, red);
  float mean = s / N_;
  float rstd = rsqrtf(s2 / N_ - mean * mean + EPS);
  const float* resr = res ? res + (size_t)bc * N_ : nullptr;
  for (int n = threadIdx.x; n < N_; n += 256) {
    float vl = (outr[n] - mean) * rstd;
    if (resr) vl += resr[n];
    outr[n] = vl >= 0.f ? vl : 0.01f * vl;
  }
}

// ------------------------------------- tokens: + pos embed + LayerNorm (both paths)
__global__ void token_ln_kernel(const float* __restrict__ x,
                                const float* __restrict__ xc,
                                const float* __restrict__ pos,
                                const float* __restrict__ lng,
                                const float* __restrict__ lnb,
                                float* __restrict__ x_n,
                                _Float16* __restrict__ x_n_h,
                                float* __restrict__ xc_n) {
  __shared__ float red[256];
  int bn = blockIdx.x;
  int b = bn / N_, n = bn % N_;
  int c = threadIdx.x;
  float pe = pos[(size_t)n * C_ + c];
  float t0 = x[((size_t)b * C_ + c) * N_ + n] + pe;
  float t1 = xc[((size_t)b * C_ + c) * N_ + n] + pe;
  float s0 = block_reduce_sum(t0, red);
  float q0 = block_reduce_sum(t0 * t0, red);
  float s1 = block_reduce_sum(t1, red);
  float q1 = block_reduce_sum(t1 * t1, red);
  float m0 = s0 / C_, m1 = s1 / C_;
  float r0 = rsqrtf(q0 / C_ - m0 * m0 + EPS);
  float r1 = rsqrtf(q1 / C_ - m1 * m1 + EPS);
  float g = lng[c], bb = lnb[c];
  float v0 = (t0 - m0) * r0 * g + bb;
  float v1 = (t1 - m1) * r1 * g + bb;
  size_t o = ((size_t)b * N_ + n) * C_ + c;
  x_n[o] = v0;
  x_n_h[o] = (_Float16)v0;
  xc_n[o] = v1;
}

// ----------------------------------------------------- QKV GEMM [BN,256]x[256,768]
// q -> f32 [bh][n][d] (needed exactly for the L2 norm); k,v -> f16 transposed
// [bh][d][N] so the K/V projection A-operand becomes contiguous f16 rows.
__global__ void qkv_kernel(const _Float16* __restrict__ xnh,
                           const _Float16* __restrict__ wqkvh,
                           float* __restrict__ qb,
                           _Float16* __restrict__ khT,
                           _Float16* __restrict__ vhT) {
  int nt = N_ / 64;
  int b = blockIdx.x / nt, tb = blockIdx.x % nt;
  int wave = threadIdx.x >> 5, lane = threadIdx.x & 31;
  int tok0 = tb * 64 + wave * 16;
  const _Float16* A = xnh + ((size_t)b * N_ + tok0) * C_;
  v16h areg[8];
#pragma unroll
  for (int kc = 0; kc < 8; ++kc) areg[kc] = load_a_h(A + kc * 32, C_, lane);
  int rbase = tok0 + ((lane < 16) ? 0 : 8);
  for (int ct = 0; ct < 48; ++ct) {
    int col0 = ct * 16;
    if (ct < 47)
      __builtin_prefetch(wqkvh + (size_t)lane * (3 * C_) + col0 + 16, 0, 1);
    v8f acc = {};
#pragma unroll
    for (int kc = 0; kc < 8; ++kc) {
      v16h bt = load_b_h(wqkvh + (size_t)(kc * 32) * (3 * C_) + col0,
                         3 * C_, lane);
      acc = wmma_f16(areg[kc], bt, acc);
    }
    int colg = col0 + (lane & 15);
    int which = colg >> 8;
    int cc = colg & 255;
    int h = cc >> 6, dd = cc & 63;
    if (which == 0) {
#pragma unroll
      for (int i = 0; i < 8; ++i)
        qb[(((size_t)b * NH_ + h) * N_ + rbase + i) * DH_ + dd] = acc[i];
    } else {
      _Float16* dstT = (which == 1) ? khT : vhT;
      size_t base = (((size_t)b * NH_ + h) * DH_ + dd) * N_ + rbase;
#pragma unroll
      for (int i = 0; i < 8; ++i) dstT[base + i] = (_Float16)acc[i];
    }
  }
}

// ------------------------------------------ 1/||q|| over token dim per (b,h,d)
__global__ void qnorm_kernel(const float* __restrict__ qb,
                             float* __restrict__ rnorm) {
  __shared__ float red[256];
  int idx = blockIdx.x;            // bh*64 + dd
  int bh = idx >> 6, dd = idx & 63;
  const float* src = qb + (size_t)bh * N_ * DH_ + dd;
  float s = 0.f;
  for (int n = threadIdx.x; n < N_; n += 256) {
    float q = src[(size_t)n * DH_];
    s += q * q;
  }
  s = block_reduce_sum(s, red);
  if (threadIdx.x == 0) rnorm[idx] = 1.f / fmaxf(sqrtf(s), 1e-12f);
}

// --------------------------- K/V low-rank projection: [d,N] x [N,P] (K-dim=13824)
// A from f16 transposed k/v -> contiguous rows, B from pre-converted f16 w_e
__global__ void kvproj_kernel(const _Float16* __restrict__ khT,
                              const _Float16* __restrict__ vhT,
                              const _Float16* __restrict__ weh,
                              const float* __restrict__ be,
                              _Float16* __restrict__ kph,     // [bh][d][p]
                              _Float16* __restrict__ vpth) {  // [bh][p][d]
  int which = blockIdx.x & 1;
  int bh = blockIdx.x >> 1;
  const _Float16* src = (which ? vhT : khT) + (size_t)bh * DH_ * N_;
  int wave = threadIdx.x >> 5, lane = threadIdx.x & 31;
  int mt = wave;                    // 16-row d-tile
  v8f acc[4] = {};
  for (int n0 = 0; n0 < N_; n0 += 32) {
    __builtin_prefetch(weh + (size_t)(n0 + 32 + lane) * P_, 0, 1);
    v16h a = load_a_h(src + (size_t)(mt * 16) * N_ + n0, N_, lane);
#pragma unroll
    for (int pt = 0; pt < 4; ++pt) {
      v16h bt = load_b_h(weh + (size_t)n0 * P_ + pt * 16, P_, lane);
      acc[pt] = wmma_f16(a, bt, acc[pt]);
    }
  }
  int rb = (lane < 16) ? 0 : 8;
#pragma unroll
  for (int pt = 0; pt < 4; ++pt) {
    int p = pt * 16 + (lane & 15);
    float bias = be[p];
#pragma unroll
    for (int i = 0; i < 8; ++i) {
      int dd = mt * 16 + rb + i;
      float vl = acc[pt][i] + bias;
      if (which == 0) kph[(size_t)bh * 4096 + dd * 64 + p]  = (_Float16)vl;
      else            vpth[(size_t)bh * 4096 + p * 64 + dd] = (_Float16)vl;
    }
  }
}

// ------------------------------------------------------- channel-attention gate
__global__ void pool_kernel(const float* __restrict__ xcn,
                            const float* __restrict__ wpool,
                            const float* __restrict__ bpool,
                            float* __restrict__ pooled) {
  __shared__ float red[256];
  int b = blockIdx.x / C_, c = blockIdx.x % C_;
  float s = 0.f;
  for (int n = threadIdx.x; n < N_; n += 256)
    s += xcn[((size_t)b * N_ + n) * C_ + c] * wpool[n];
  s = block_reduce_sum(s, red);
  if (threadIdx.x == 0) pooled[blockIdx.x] = s + bpool[0];
}

__global__ void gate_kernel(const float* __restrict__ pooled,
                            const float* __restrict__ wfc1,
                            const float* __restrict__ wfc2,
                            float* __restrict__ gate) {
  __shared__ float pl[C_];
  __shared__ float hid[DH_];
  int b = blockIdx.x, tid = threadIdx.x;
  pl[tid] = pooled[b * C_ + tid];
  __syncthreads();
  if (tid < DH_) {
    float s = 0.f;
    for (int c = 0; c < C_; ++c) s += pl[c] * wfc1[c * DH_ + tid];
    hid[tid] = fmaxf(s, 0.f);
  }
  __syncthreads();
  float s = 0.f;
  for (int j = 0; j < DH_; ++j) s += hid[j] * wfc2[j * C_ + tid];
  gate[b * C_ + tid] = 1.f / (1.f + expf(-s));
}

// ------- fused attention: LDS-staged K_p/V_p^T, WMMA -> softmax(P=64) -> WMMA
__global__ void attn_kernel(const float* __restrict__ qb,
                            const _Float16* __restrict__ kph,
                            const _Float16* __restrict__ vpth,
                            const float* __restrict__ rnorm,
                            const float* __restrict__ temp2,
                            _Float16* __restrict__ sah) {
  __shared__ float    attnS[4][16][64];
  __shared__ _Float16 attnH[4][16][64];
  __shared__ __align__(16) _Float16 kpS[4096];
  __shared__ __align__(16) _Float16 vptS[4096];
  __shared__ float    rns[64];
  int nt = N_ / 64;
  int bh = blockIdx.x / nt, tb = blockIdx.x % nt;
  int b = bh >> 2, h = bh & 3;
  int wave = threadIdx.x >> 5, lane = threadIdx.x & 31;
  int tok0 = tb * 64 + wave * 16;
  // stage K_p [d][p] and V_p^T [p][d] (8 KB each) into LDS once per block
  {
    const uint4* gk = (const uint4*)(kph  + (size_t)bh * 4096);
    const uint4* gv = (const uint4*)(vpth + (size_t)bh * 4096);
    uint4* sk = (uint4*)kpS;
    uint4* sv = (uint4*)vptS;
    for (int i = threadIdx.x; i < 512; i += 128) { sk[i] = gk[i]; sv[i] = gv[i]; }
  }
  if (threadIdx.x < 64) rns[threadIdx.x] = rnorm[bh * 64 + threadIdx.x];
  __syncthreads();
  const float* q = qb + (size_t)bh * N_ * DH_;
  int rb = (lane < 16) ? 0 : 8;

  v8f acc[4] = {};
#pragma unroll
  for (int kc = 0; kc < 2; ++kc) {
    v16h a = load_a_gen([&](int m, int k) {
      int dd = kc * 32 + k;
      return q[(size_t)(tok0 + m) * DH_ + dd] * rns[dd]; }, lane);
#pragma unroll
    for (int pt = 0; pt < 4; ++pt) {
      v16h bt = load_b_h(kpS + (kc * 32) * 64 + pt * 16, 64, lane);
      acc[pt] = wmma_f16(a, bt, acc[pt]);
    }
  }
  float tmp = temp2[h];
#pragma unroll
  for (int pt = 0; pt < 4; ++pt)
#pragma unroll
    for (int i = 0; i < 8; ++i)
      attnS[wave][rb + i][pt * 16 + (lane & 15)] = acc[pt][i] * tmp;
  __syncthreads();

  if (lane < 16) {               // one row (token) per lane; softmax over P=64
    float* row = attnS[wave][lane];
    float mx = -3.4e38f;
    for (int p = 0; p < 64; ++p) mx = fmaxf(mx, row[p]);
    float sm = 0.f;
    for (int p = 0; p < 64; ++p) { float e = expf(row[p] - mx); row[p] = e; sm += e; }
    float inv = 1.f / sm;
    for (int p = 0; p < 64; ++p) attnH[wave][lane][p] = (_Float16)(row[p] * inv);
  }
  __syncthreads();

  v8f acc2[4] = {};
#pragma unroll
  for (int kc = 0; kc < 2; ++kc) {
    v16h a2 = load_a_h(&attnH[wave][0][0] + kc * 32, 64, lane);
#pragma unroll
    for (int dt = 0; dt < 4; ++dt) {
      v16h bt = load_b_h(vptS + (kc * 32) * 64 + dt * 16, 64, lane);
      acc2[dt] = wmma_f16(a2, bt, acc2[dt]);
    }
  }
  // store f16 into the reference's permute(0,3,1,2).reshape scramble: [B][d][h][N]
#pragma unroll
  for (int dt = 0; dt < 4; ++dt) {
    int dd = dt * 16 + (lane & 15);
    size_t base = ((size_t)(b * 64 + dd) * NH_ + h) * N_ + tok0 + rb;
#pragma unroll
    for (int i = 0; i < 8; ++i) sah[base + i] = (_Float16)acc2[dt][i];
  }
}

// ------- out projections + bias + gamma residual + final [B,N,C]->[B,C,N] transpose
__global__ void outproj_kernel(const _Float16* __restrict__ sah,
                               const float* __restrict__ xcn,
                               const float* __restrict__ gate,
                               const float* __restrict__ xn,
                               const _Float16* __restrict__ w1h,
                               const float* __restrict__ bo1,
                               const _Float16* __restrict__ w2h,
                               const float* __restrict__ bo2,
                               const float* __restrict__ gamma,
                               float* __restrict__ out) {
  int nt = N_ / 64;
  int b = blockIdx.x / nt, tb = blockIdx.x % nt;
  int wave = threadIdx.x >> 5, lane = threadIdx.x & 31;
  int tok0 = tb * 64 + wave * 16;
  int rb = (lane < 16) ? 0 : 8;
  v16h areg[8];

  // GEMM 1: x_SA(f16, scrambled row-major [B][N][C]) @ w_out1 -> cols 0..127
#pragma unroll
  for (int kc = 0; kc < 8; ++kc)
    areg[kc] = load_a_h(sah + (size_t)b * C_ * N_ + (size_t)tok0 * C_ + kc * 32,
                        C_, lane);
  for (int ct = 0; ct < 8; ++ct) {
    v8f acc = {};
#pragma unroll
    for (int kc = 0; kc < 8; ++kc) {
      v16h bt = load_b_h(w1h + (size_t)(kc * 32) * 128 + ct * 16, 128, lane);
      acc = wmma_f16(areg[kc], bt, acc);
    }
    int c = ct * 16 + (lane & 15);
    float bias = bo1[c], gm = gamma[c];
#pragma unroll
    for (int i = 0; i < 8; ++i) {
      int tok = tok0 + rb + i;
      out[((size_t)b * C_ + c) * N_ + tok] =
          xn[((size_t)b * N_ + tok) * C_ + c] + gm * (acc[i] + bias);
    }
  }

  // GEMM 2: (xc_n * gate) @ w_out2 -> cols 128..255
#pragma unroll
  for (int kc = 0; kc < 8; ++kc)
    areg[kc] = load_a_gen([&](int m, int k) {
      int cc = kc * 32 + k;
      return xcn[((size_t)b * N_ + tok0 + m) * C_ + cc] * gate[b * C_ + cc]; }, lane);
  for (int ct = 0; ct < 8; ++ct) {
    v8f acc = {};
#pragma unroll
    for (int kc = 0; kc < 8; ++kc) {
      v16h bt = load_b_h(w2h + (size_t)(kc * 32) * 128 + ct * 16, 128, lane);
      acc = wmma_f16(areg[kc], bt, acc);
    }
    int c = 128 + ct * 16 + (lane & 15);
    float bias = bo2[c - 128], gm = gamma[c];
#pragma unroll
    for (int i = 0; i < 8; ++i) {
      int tok = tok0 + rb + i;
      out[((size_t)b * C_ + c) * N_ + tok] =
          xn[((size_t)b * N_ + tok) * C_ + c] + gm * (acc[i] + bias);
    }
  }
}

// ---------------------------------------------------------------------- launch
extern "C" void kernel_launch(void* const* d_in, const int* in_sizes, int n_in,
                              void* d_out, int out_size, void* d_ws, size_t ws_size,
                              hipStream_t stream) {
  (void)in_sizes; (void)n_in; (void)out_size; (void)ws_size;
  const float* x     = (const float*)d_in[0];
  const float* pos   = (const float*)d_in[1];
  const float* lng   = (const float*)d_in[2];
  const float* lnb   = (const float*)d_in[3];
  const float* gamma = (const float*)d_in[4];
  const float* temp2 = (const float*)d_in[5];
  const float* wqkv  = (const float*)d_in[6];
  const float* we    = (const float*)d_in[7];
  const float* be    = (const float*)d_in[8];
  const float* wpool = (const float*)d_in[9];
  const float* bpool = (const float*)d_in[10];
  const float* wfc1  = (const float*)d_in[11];
  const float* wfc2  = (const float*)d_in[12];
  const float* w1    = (const float*)d_in[13];
  const float* bo1   = (const float*)d_in[14];
  const float* w2    = (const float*)d_in[15];
  const float* bo2   = (const float*)d_in[16];
  const float* wc1   = (const float*)d_in[17];
  const float* bc1   = (const float*)d_in[18];
  const float* wc2   = (const float*)d_in[19];
  const float* bc2   = (const float*)d_in[20];
  float* out = (float*)d_out;

  const size_t BCN = (size_t)B_ * C_ * N_;
  char* p = (char*)d_ws;
  auto alloc = [&](size_t bytes) {
    char* r = p; p += (bytes + 255) & ~(size_t)255; return r; };
  float*    h1     = (float*)alloc(BCN * 4);
  float*    xc     = (float*)alloc(BCN * 4);
  float*    x_n    = (float*)alloc(BCN * 4);
  float*    xc_n   = (float*)alloc(BCN * 4);
  float*    qb     = (float*)alloc(BCN * 4);
  _Float16* xnh    = (_Float16*)alloc(BCN * 2);
  _Float16* khT    = (_Float16*)alloc(BCN * 2);
  _Float16* vhT    = (_Float16*)alloc(BCN * 2);
  _Float16* sah    = (_Float16*)alloc(BCN * 2);
  _Float16* kph    = (_Float16*)alloc((size_t)B_ * NH_ * 4096 * 2);
  _Float16* vpth   = (_Float16*)alloc((size_t)B_ * NH_ * 4096 * 2);
  _Float16* wqkvh  = (_Float16*)alloc((size_t)C_ * 3 * C_ * 2);
  _Float16* weh    = (_Float16*)alloc((size_t)N_ * P_ * 2);
  _Float16* w1h    = (_Float16*)alloc((size_t)C_ * 128 * 2);
  _Float16* w2h    = (_Float16*)alloc((size_t)C_ * 128 * 2);
  float*    rnorm  = (float*)alloc((size_t)B_ * NH_ * DH_ * 4);
  float*    pooled = (float*)alloc((size_t)B_ * C_ * 4);
  float*    gate   = (float*)alloc((size_t)B_ * C_ * 4);

  int nt = N_ / 64;
  // weight pre-conversion (once per call, tiny)
  cvt_f16_kernel<<<(C_*3*C_ + 255)/256, 256, 0, stream>>>(wqkv, wqkvh, C_*3*C_);
  cvt_f16_kernel<<<(N_*P_  + 255)/256, 256, 0, stream>>>(we,   weh,   N_*P_);
  cvt_f16_kernel<<<(C_*128 + 255)/256, 256, 0, stream>>>(w1,   w1h,   C_*128);
  cvt_f16_kernel<<<(C_*128 + 255)/256, 256, 0, stream>>>(w2,   w2h,   C_*128);

  gconv_inorm_kernel<<<B_ * C_, 256, 0, stream>>>(x,  wc1, bc1, nullptr, h1);
  gconv_inorm_kernel<<<B_ * C_, 256, 0, stream>>>(h1, wc2, bc2, x,       xc);
  token_ln_kernel<<<B_ * N_, 256, 0, stream>>>(x, xc, pos, lng, lnb, x_n, xnh, xc_n);
  qkv_kernel<<<B_ * nt, 128, 0, stream>>>(xnh, wqkvh, qb, khT, vhT);
  qnorm_kernel<<<B_ * NH_ * DH_, 256, 0, stream>>>(qb, rnorm);
  kvproj_kernel<<<B_ * NH_ * 2, 128, 0, stream>>>(khT, vhT, weh, be, kph, vpth);
  pool_kernel<<<B_ * C_, 256, 0, stream>>>(xc_n, wpool, bpool, pooled);
  gate_kernel<<<B_, 256, 0, stream>>>(pooled, wfc1, wfc2, gate);
  attn_kernel<<<B_ * NH_ * nt, 128, 0, stream>>>(qb, kph, vpth, rnorm, temp2, sah);
  outproj_kernel<<<B_ * nt, 128, 0, stream>>>(sah, xc_n, gate, x_n,
                                              w1h, bo1, w2h, bo2, gamma, out);
}
